// BigramHash_25228637897406
// MI455X (gfx1250) — matrix operand order repaired and checked by reading, not compiled, not run on hardware
//
#include <hip/hip_runtime.h>
#include <hip/hip_bf16.h>

// ---------------------------------------------------------------------------
// BigramHash: out[M,N] = tab[hash(t)] @ w^T,  M=32768, N=K=1024, all f32.
// MI455X strategy:
//   1) prepass splits tab/w once into bf16 hi/lo planes in d_ws (16 MB, L2-
//      resident) -> kills ~1G redundant per-WG f32->bf16 conversions.
//   2) bf16x3 split WMMA GEMM (hi*hi + hi*lo + lo*hi), f32 accumulation.
//      Tiles staged global->LDS with gfx1250 async-to-LDS ops (ASYNCcnt),
//      per-lane addressed to implement the hashed row gather.
// HBM traffic ~= 150MB (output-dominated); K-loop is wmma/ds/async only.
// ---------------------------------------------------------------------------

#define SZc 3072
#define Dc  1024
#define Tc  8192
#define Mc  32768

#define TM 128      // workgroup M tile
#define TN 128      // workgroup N tile
#define TK 32       // K step = one bf16 WMMA depth
#define LDP 40      // padded LDS row stride in halves (80 bytes, 16B-aligned)

typedef __attribute__((ext_vector_type(16))) __bf16       v16bf;
typedef __attribute__((ext_vector_type(8)))  float        v8f;
typedef __attribute__((ext_vector_type(4)))  float        v4f;
typedef __attribute__((ext_vector_type(4)))  unsigned int v4u;
typedef __attribute__((ext_vector_type(2)))  unsigned int v2u;

struct Q32 { v4u a, b; };   // 32 bytes -> bit_cast to v16bf

static __device__ __forceinline__ unsigned short bf16_rne(float f) {
  unsigned int u = __builtin_bit_cast(unsigned int, f);
  u += 0x7FFFu + ((u >> 16) & 1u);
  return (unsigned short)(u >> 16);
}
static __device__ __forceinline__ float bf16_f32(unsigned short h) {
  return __builtin_bit_cast(float, (unsigned int)h << 16);
}

// Low 32 bits of a flat pointer into LDS == wave-relative LDS byte offset
// (flat->LDS aperture mapping truncates to addr[31:0]).
static __device__ __forceinline__ unsigned int lds_off(const void* p) {
  return (unsigned int)(unsigned long long)p;
}

// gfx1250 async global->LDS copy, 16B per lane, tracked by ASYNCcnt.
static __device__ __forceinline__ void async_ld_b128(const void* gaddr,
                                                     unsigned int loff) {
  asm volatile("global_load_async_to_lds_b128 %0, %1, off"
               :: "v"(loff), "v"(gaddr) : "memory");
}
static __device__ __forceinline__ void wait_async0() {
#if __has_builtin(__builtin_amdgcn_s_wait_asynccnt)
  __builtin_amdgcn_s_wait_asynccnt(0);
#else
  asm volatile("s_wait_asynccnt 0x0" ::: "memory");
#endif
}

// ---------------------------------------------------------------------------
// Prepass: split f32 -> (bf16 hi, bf16 lo) planes for tab and w.
// 4,194,304 elements total, 4 per thread.
// ---------------------------------------------------------------------------
__global__ __launch_bounds__(256)
void split_prepass(const float* __restrict__ tab, const float* __restrict__ w,
                   unsigned short* __restrict__ tabH,
                   unsigned short* __restrict__ tabL,
                   unsigned short* __restrict__ wH,
                   unsigned short* __restrict__ wL) {
  const int NT = SZc * Dc;                     // 3145728
  int i = (blockIdx.x * 256 + threadIdx.x) * 4;
  const float* src;
  unsigned short *dh, *dl;
  int j;
  if (i < NT) { src = tab; dh = tabH; dl = tabL; j = i; }
  else        { src = w;   dh = wH;   dl = wL;   j = i - NT; }
  v4f x = *(const v4f*)(src + j);
  unsigned short h[4], l[4];
  #pragma unroll
  for (int e = 0; e < 4; ++e) {
    h[e] = bf16_rne(x[e]);
    l[e] = bf16_rne(x[e] - bf16_f32(h[e]));
  }
  v2u hp, lp;
  hp[0] = (unsigned)h[0] | ((unsigned)h[1] << 16);
  hp[1] = (unsigned)h[2] | ((unsigned)h[3] << 16);
  lp[0] = (unsigned)l[0] | ((unsigned)l[1] << 16);
  lp[1] = (unsigned)l[2] | ((unsigned)l[3] << 16);
  *(v2u*)(dh + j) = hp;
  *(v2u*)(dl + j) = lp;
}

// ---------------------------------------------------------------------------
// Main GEMM: async-staged bf16 planes, 24 WMMAs per K-step per wave.
// ---------------------------------------------------------------------------
__global__ __launch_bounds__(256)
void bigram_proj_wmma_async(const int* __restrict__ t,
                            const unsigned short* __restrict__ tabH,
                            const unsigned short* __restrict__ tabL,
                            const unsigned short* __restrict__ wH,
                            const unsigned short* __restrict__ wL,
                            float* __restrict__ out) {
  __shared__ unsigned short sAh[TM * LDP];
  __shared__ unsigned short sAl[TM * LDP];
  __shared__ unsigned short sBh[TN * LDP];
  __shared__ unsigned short sBl[TN * LDP];

  const int tid  = threadIdx.x;
  const int lane = tid & 31;
  const int wv   = tid >> 5;       // wave 0..7
  const int wm   = wv >> 1;        // 0..3 : wave M slot (32 rows)
  const int wn   = wv & 1;         // 0..1 : wave N slot (64 cols)
  const int h    = (lane >> 4) & 1;
  const int ln   = lane & 15;

  const int tm = blockIdx.y * TM;
  const int tn = blockIdx.x * TN;

  // ---- A loader: 2 threads per row, hash hoisted out of K loop ----
  const int arow  = tid >> 1;      // 0..127
  const int ahalf = tid & 1;       // 16-half (32B) chunk of the 32-half row
  const int gm = tm + arow;
  const int tok  = t[gm];
  const int prev = ((gm & (Tc - 1)) == 0) ? 0 : t[gm - 1];
  const unsigned int aidx =
      ((unsigned int)(tok % SZc) * 617u +          // 31337  % 3072
       (unsigned int)(prev % SZc) * 1603u)         // 1000003 % 3072
      % (unsigned int)SZc;
  const unsigned short* aH = tabH + (size_t)aidx * Dc + ahalf * 16;
  const unsigned short* aL = tabL + (size_t)aidx * Dc + ahalf * 16;

  // ---- B loader: 2 threads per row; w rows staged un-transposed:
  //      B[k][n] = w[n][k] == the WMMA B fragment's 32 contiguous bytes ----
  const int brow  = tid >> 1;      // 0..127
  const int bhalf = tid & 1;
  const unsigned short* bH = wH + (size_t)(tn + brow) * Dc + bhalf * 16;
  const unsigned short* bL = wL + (size_t)(tn + brow) * Dc + bhalf * 16;

  const unsigned int aHo = lds_off(&sAh[arow * LDP + ahalf * 16]);
  const unsigned int aLo = lds_off(&sAl[arow * LDP + ahalf * 16]);
  const unsigned int bHo = lds_off(&sBh[brow * LDP + bhalf * 16]);
  const unsigned int bLo = lds_off(&sBl[brow * LDP + bhalf * 16]);

  v8f acc[2][4] = {};

  for (int kk = 0; kk < Dc; kk += TK) {
    if (kk + TK < Dc) {            // warm L2 for next step (global_prefetch_b8)
      __builtin_prefetch(aH + kk + TK, 0, 1);
      __builtin_prefetch(bH + kk + TK, 0, 1);
    }
    __syncthreads();               // previous iteration's compute done

    // ---- async global -> LDS staging: 8 x B128, no VALU, no VGPR data ----
    async_ld_b128(aH + kk,     aHo);
    async_ld_b128(aH + kk + 8, aHo + 16);
    async_ld_b128(aL + kk,     aLo);
    async_ld_b128(aL + kk + 8, aLo + 16);
    async_ld_b128(bH + kk,     bHo);
    async_ld_b128(bH + kk + 8, bHo + 16);
    async_ld_b128(bL + kk,     bLo);
    async_ld_b128(bL + kk + 8, bLo + 16);
    wait_async0();
    __syncthreads();               // all waves' tiles staged

    // ---- fragments in contiguous k order (k = 16h + s), same for A and B ----
    // A (16x32 bf16): lane m holds 16 consecutive K at offset 16h
    v16bf ah[2], al_[2], bh[4], bl[4];
    #pragma unroll
    for (int mi = 0; mi < 2; ++mi) {
      const int r = wm * 32 + mi * 16 + ln;
      const unsigned short* p  = &sAh[r * LDP];
      const unsigned short* pl = &sAl[r * LDP];
      Q32 q, ql;
      q.a  = *(const v4u*)(p + 16 * h);
      q.b  = *(const v4u*)(p + 16 * h + 8);
      ql.a = *(const v4u*)(pl + 16 * h);
      ql.b = *(const v4u*)(pl + 16 * h + 8);
      ah[mi]  = __builtin_bit_cast(v16bf, q);
      al_[mi] = __builtin_bit_cast(v16bf, ql);
    }
    // B (32x16 bf16): lane n holds 16 consecutive K at offset 16h
    #pragma unroll
    for (int ni = 0; ni < 4; ++ni) {
      const int r = wn * 64 + ni * 16 + ln;
      const unsigned short* p  = &sBh[r * LDP];
      const unsigned short* pl = &sBl[r * LDP];
      Q32 q, ql;
      q.a  = *(const v4u*)(p + 16 * h);
      q.b  = *(const v4u*)(p + 16 * h + 8);
      ql.a = *(const v4u*)(pl + 16 * h);
      ql.b = *(const v4u*)(pl + 16 * h + 8);
      bh[ni] = __builtin_bit_cast(v16bf, q);
      bl[ni] = __builtin_bit_cast(v16bf, ql);
    }

    // ---- 24 WMMAs per K step: hi*hi + hi*lo + lo*hi ----
    #pragma unroll
    for (int mi = 0; mi < 2; ++mi)
      #pragma unroll
      for (int ni = 0; ni < 4; ++ni) {
        acc[mi][ni] = __builtin_amdgcn_wmma_f32_16x16x32_bf16(
            false, ah[mi],  false, bh[ni], (short)0, acc[mi][ni], false, false);
        acc[mi][ni] = __builtin_amdgcn_wmma_f32_16x16x32_bf16(
            false, ah[mi],  false, bl[ni], (short)0, acc[mi][ni], false, false);
        acc[mi][ni] = __builtin_amdgcn_wmma_f32_16x16x32_bf16(
            false, al_[mi], false, bh[ni], (short)0, acc[mi][ni], false, false);
      }
  }

  // ---- epilogue: C/D layout = (m = v + 8h, n = lane&15) ----
  #pragma unroll
  for (int mi = 0; mi < 2; ++mi) {
    const int mbase = tm + wm * 32 + mi * 16 + 8 * h;
    #pragma unroll
    for (int ni = 0; ni < 4; ++ni) {
      const int n = tn + wn * 64 + ni * 16 + ln;
      #pragma unroll
      for (int v = 0; v < 8; ++v) {
        out[(size_t)(mbase + v) * Dc + n] = acc[mi][ni][v];
      }
    }
  }
}

// ---------------------------------------------------------------------------
// Fallback (ws too small): split inline each K-step (round-2 kernel).
// ---------------------------------------------------------------------------
__global__ __launch_bounds__(256)
void bigram_proj_wmma_inline(const int* __restrict__ t,
                             const float* __restrict__ tab,
                             const float* __restrict__ w,
                             float* __restrict__ out) {
  __shared__ unsigned short sAh[TM * LDP];
  __shared__ unsigned short sAl[TM * LDP];
  __shared__ unsigned short sBh[TN * LDP];
  __shared__ unsigned short sBl[TN * LDP];

  const int tid  = threadIdx.x;
  const int lane = tid & 31;
  const int wv   = tid >> 5;
  const int wm   = wv >> 1;
  const int wn   = wv & 1;
  const int h    = (lane >> 4) & 1;
  const int ln   = lane & 15;
  const int tm = blockIdx.y * TM;
  const int tn = blockIdx.x * TN;

  const int arow  = tid >> 1;
  const int ahalf = tid & 1;
  const int gm = tm + arow;
  const int tok  = t[gm];
  const int prev = ((gm & (Tc - 1)) == 0) ? 0 : t[gm - 1];
  const unsigned int aidx =
      ((unsigned int)(tok % SZc) * 617u +
       (unsigned int)(prev % SZc) * 1603u) % (unsigned int)SZc;
  const float* arow_ptr = tab + (size_t)aidx * Dc + ahalf * 16;
  const int brow  = tid >> 1;
  const int bhalf = tid & 1;
  const float* brow_ptr = w + (size_t)(tn + brow) * Dc + bhalf * 16;

  v8f acc[2][4] = {};

  for (int kk = 0; kk < Dc; kk += TK) {
    v4f av[4], bv[4];
    #pragma unroll
    for (int i = 0; i < 4; ++i) av[i] = *(const v4f*)(arow_ptr + kk + 4 * i);
    #pragma unroll
    for (int i = 0; i < 4; ++i) bv[i] = *(const v4f*)(brow_ptr + kk + 4 * i);

    __syncthreads();
    {
      const float* af = (const float*)av;
      unsigned short uh[16], ul[16];
      #pragma unroll
      for (int i = 0; i < 16; ++i) {
        uh[i] = bf16_rne(af[i]);
        ul[i] = bf16_rne(af[i] - bf16_f32(uh[i]));
      }
      unsigned short* dh = &sAh[arow * LDP + ahalf * 16];
      unsigned short* dl = &sAl[arow * LDP + ahalf * 16];
      *(v4u*)(dh)     = *(const v4u*)(&uh[0]);
      *(v4u*)(dh + 8) = *(const v4u*)(&uh[8]);
      *(v4u*)(dl)     = *(const v4u*)(&ul[0]);
      *(v4u*)(dl + 8) = *(const v4u*)(&ul[8]);
    }
    {
      const float* bf = (const float*)bv;
      unsigned short uh[16], ul[16];
      #pragma unroll
      for (int i = 0; i < 16; ++i) {
        uh[i] = bf16_rne(bf[i]);
        ul[i] = bf16_rne(bf[i] - bf16_f32(uh[i]));
      }
      unsigned short* dh = &sBh[brow * LDP + bhalf * 16];
      unsigned short* dl = &sBl[brow * LDP + bhalf * 16];
      *(v4u*)(dh)     = *(const v4u*)(&uh[0]);
      *(v4u*)(dh + 8) = *(const v4u*)(&uh[8]);
      *(v4u*)(dl)     = *(const v4u*)(&ul[0]);
      *(v4u*)(dl + 8) = *(const v4u*)(&ul[8]);
    }
    __syncthreads();

    v16bf ah[2], al_[2], bh[4], bl[4];
    #pragma unroll
    for (int mi = 0; mi < 2; ++mi) {
      const int r = wm * 32 + mi * 16 + ln;
      const unsigned short* p  = &sAh[r * LDP];
      const unsigned short* pl = &sAl[r * LDP];
      Q32 q, ql;
      q.a  = *(const v4u*)(p + 16 * h);
      q.b  = *(const v4u*)(p + 16 * h + 8);
      ql.a = *(const v4u*)(pl + 16 * h);
      ql.b = *(const v4u*)(pl + 16 * h + 8);
      ah[mi]  = __builtin_bit_cast(v16bf, q);
      al_[mi] = __builtin_bit_cast(v16bf, ql);
    }
    #pragma unroll
    for (int ni = 0; ni < 4; ++ni) {
      const int r = wn * 64 + ni * 16 + ln;
      const unsigned short* p  = &sBh[r * LDP];
      const unsigned short* pl = &sBl[r * LDP];
      Q32 q, ql;
      q.a  = *(const v4u*)(p + 16 * h);
      q.b  = *(const v4u*)(p + 16 * h + 8);
      ql.a = *(const v4u*)(pl + 16 * h);
      ql.b = *(const v4u*)(pl + 16 * h + 8);
      bh[ni] = __builtin_bit_cast(v16bf, q);
      bl[ni] = __builtin_bit_cast(v16bf, ql);
    }
    #pragma unroll
    for (int mi = 0; mi < 2; ++mi)
      #pragma unroll
      for (int ni = 0; ni < 4; ++ni) {
        acc[mi][ni] = __builtin_amdgcn_wmma_f32_16x16x32_bf16(
            false, ah[mi],  false, bh[ni], (short)0, acc[mi][ni], false, false);
        acc[mi][ni] = __builtin_amdgcn_wmma_f32_16x16x32_bf16(
            false, ah[mi],  false, bl[ni], (short)0, acc[mi][ni], false, false);
        acc[mi][ni] = __builtin_amdgcn_wmma_f32_16x16x32_bf16(
            false, al_[mi], false, bh[ni], (short)0, acc[mi][ni], false, false);
      }
  }

  #pragma unroll
  for (int mi = 0; mi < 2; ++mi) {
    const int mbase = tm + wm * 32 + mi * 16 + 8 * h;
    #pragma unroll
    for (int ni = 0; ni < 4; ++ni) {
      const int n = tn + wn * 64 + ni * 16 + ln;
      #pragma unroll
      for (int v = 0; v < 8; ++v) {
        out[(size_t)(mbase + v) * Dc + n] = acc[mi][ni][v];
      }
    }
  }
}

extern "C" void kernel_launch(void* const* d_in, const int* in_sizes, int n_in,
                              void* d_out, int out_size, void* d_ws, size_t ws_size,
                              hipStream_t stream) {
  const int*   t   = (const int*)d_in[0];
  const float* tab = (const float*)d_in[1];
  const float* w   = (const float*)d_in[2];
  float*       out = (float*)d_out;

  dim3 grid(Dc / TN, Mc / TM);   // (8, 256)
  dim3 block(256);

  const size_t need = ((size_t)SZc * Dc + (size_t)Dc * Dc) * 2 /*planes*/
                      * sizeof(unsigned short);   // 16 MB
  if (ws_size >= need) {
    unsigned short* tabH = (unsigned short*)d_ws;
    unsigned short* tabL = tabH + (size_t)SZc * Dc;
    unsigned short* wH   = tabL + (size_t)SZc * Dc;
    unsigned short* wL   = wH + (size_t)Dc * Dc;
    const int total_quads = (SZc * Dc + Dc * Dc) / 4;       // 1048576
    split_prepass<<<total_quads / 256, 256, 0, stream>>>(tab, w, tabH, tabL,
                                                         wH, wL);
    bigram_proj_wmma_async<<<grid, block, 0, stream>>>(t, tabH, tabL, wH, wL,
                                                       out);
  } else {
    bigram_proj_wmma_inline<<<grid, block, 0, stream>>>(t, tab, w, out);
  }
}
